// DinoFusion_18554258719349
// MI455X (gfx1250) — matrix operand order repaired and checked by reading, not compiled, not run on hardware
//
#include <hip/hip_runtime.h>
#include <hip/hip_bf16.h>
#include <math.h>

// ---------------------------------------------------------------------------
// Types for CDNA5 WMMA (gfx1250, wave32)
// ---------------------------------------------------------------------------
typedef _Float16 h16;
typedef __attribute__((ext_vector_type(16))) _Float16 v16h;
typedef __attribute__((ext_vector_type(8)))  float    v8f;
typedef int v2i_t __attribute__((vector_size(8)));

// Problem constants (from reference)
#define BB   4
#define HH   128
#define WW   128
#define NN   (HH * WW)       // 16384
#define CC   384
#define HEADS 8
#define CHD   48
#define HID   768
#define FFN2  1536

// ---------------------------------------------------------------------------
// WMMA helpers
// ---------------------------------------------------------------------------
__device__ __forceinline__ v8f wmma_f16(v16h a, v16h b, v8f c) {
  return __builtin_amdgcn_wmma_f32_16x16x32_f16(
      false, a, false, b, (short)0, c, false, false);
}

// A fragment (16x32 f16, MxK) from row-major matrix, ld in halves.
// lane L: row M = L&15 ; element e -> K = (L>>4)*16 + e
// -> two contiguous 16B runs per lane (compiler emits 2x global_load_b128)
__device__ __forceinline__ v16h load_A_rowmajor(const h16* __restrict__ base, int ld) {
  int lane = threadIdx.x & 31;
  const h16* p = base + (size_t)(lane & 15) * ld + ((lane >> 4) << 4);
  v16h a;
#pragma unroll
  for (int pr = 0; pr < 8; ++pr) {
    int off = 2 * pr;
    a[2 * pr]     = p[off];
    a[2 * pr + 1] = p[off + 1];
  }
  return a;
}

// B fragment (32x16 f16, KxN) where B[K][j] == M[j][K], M row-major with ld.
// lane L: col j = L&15, elements e -> K = (L>>4)*16 + e : one contiguous
// 32B run per lane (2x b128).
__device__ __forceinline__ v16h load_B_rows(const h16* __restrict__ base, int ld) {
  int lane = threadIdx.x & 31;
  const h16* p = base + (size_t)(lane & 15) * ld + ((lane >> 4) << 4);
  return *(const v16h*)p;
}

// ---------------------------------------------------------------------------
// Weight convert f32 -> f16
// ---------------------------------------------------------------------------
__global__ void df_f32_to_f16(const float* __restrict__ src, h16* __restrict__ dst, int n) {
  int i = blockIdx.x * 256 + threadIdx.x;
  if (i < n) dst[i] = (h16)src[i];
}

// ---------------------------------------------------------------------------
// Tiled transpose f16: in [R][Cl] -> out [Cl][R], per batch.
// Both sides coalesced; optional async global->LDS staging (CDNA5).
// ---------------------------------------------------------------------------
__global__ __launch_bounds__(256) void df_transpose16(
    const h16* __restrict__ in, h16* __restrict__ out,
    int R, int Cl, size_t inBatchStride, size_t outBatchStride) {
  __shared__ h16 s[32][40];   // 80B row stride: 8B/16B aligned chunks
  int b = blockIdx.z;
  const h16* ib = in + (size_t)b * inBatchStride;
  h16* ob = out + (size_t)b * outBatchStride;
  int ct = blockIdx.x * 32;   // column tile (fast dim of input)
  int rt = blockIdx.y * 32;   // row tile

  int tid = threadIdx.x;
  int row = tid >> 3;              // 0..31
  int chk = (tid & 7) * 4;         // 0,4,..,28 halves (8B chunks)
  const h16* gsrc = ib + (size_t)(rt + row) * Cl + ct + chk;
  h16* ldst = &s[row][chk];

#if defined(__HIP_DEVICE_COMPILE__) && __has_builtin(__builtin_amdgcn_global_load_async_to_lds_b64)
  __builtin_amdgcn_global_load_async_to_lds_b64(
      (__attribute__((address_space(1))) v2i_t*)gsrc,
      (__attribute__((address_space(3))) v2i_t*)ldst, 0, 0);
#if __has_builtin(__builtin_amdgcn_s_wait_asynccnt)
  __builtin_amdgcn_s_wait_asynccnt(0);
#else
  asm volatile("s_wait_asynccnt 0" ::: "memory");
#endif
#else
  *(uint2*)ldst = *(const uint2*)gsrc;
#endif
  __syncthreads();

  int tx = tid & 31, ty = tid >> 5;   // ty 0..7
#pragma unroll
  for (int i = 0; i < 4; ++i) {
    int r = ty + 8 * i;
    ob[(size_t)(ct + r) * R + rt + tx] = s[tx][r];
  }
}

// ---------------------------------------------------------------------------
// Channel LayerNorm (over C) of f32 [B,C,N] -> f16 [B,C,N]
// ---------------------------------------------------------------------------
__global__ void df_layernorm_f16(const float* __restrict__ x,
                                 const float* __restrict__ gam,
                                 const float* __restrict__ bet,
                                 h16* __restrict__ out, int C, int N) {
  int col = blockIdx.x * 256 + threadIdx.x;      // b*N + n
  int b = col / N;
  int n = col - b * N;
  const float* xb = x + (size_t)b * C * N + n;
  float s = 0.f, s2 = 0.f;
  for (int c = 0; c < C; ++c) {
    float v = xb[(size_t)c * N];
    s += v; s2 += v * v;
  }
  float mu  = s / (float)C;
  float var = s2 / (float)C - mu * mu;
  float inv = rsqrtf(var + 1e-5f);
  h16* ob = out + (size_t)b * C * N + n;
  for (int c = 0; c < C; ++c) {
    float v = (xb[(size_t)c * N] - mu) * inv * gam[c] + bet[c];
    ob[(size_t)c * N] = (h16)v;
  }
}

// ---------------------------------------------------------------------------
// Channel GEMM, all-contiguous WMMA operands (no LDS):
//   Y[b][o][n] = sum_k W[o][k] * Xt[b][n][k]   (+ optional f32 residual)
// Wave tile 32(O) x 64(N); block 8 waves = 128(O) x 128(N).
// ---------------------------------------------------------------------------
__global__ __launch_bounds__(256) void df_gemm_wmma_t(
    const h16*  __restrict__ Wgt,    // [O][K] f16
    const h16*  __restrict__ Xt,     // [B][N][K] f16 (N-major)
    const float* __restrict__ resid, // [B][O][N] f32 or nullptr
    float* __restrict__ outF,        // [B][O][N] f32 or nullptr
    h16*   __restrict__ outH,        // [B][O][N] f16 or nullptr
    int O, int K, int N) {
  int b = blockIdx.z;
  const h16* Xb = Xt + (size_t)b * N * K;

  int wave = threadIdx.x >> 5;
  int lane = threadIdx.x & 31;
  int wm = wave & 3;                         // 4 waves over O
  int wn = wave >> 2;                        // 2 waves over N
  int oBase = blockIdx.y * 128 + wm * 32;
  int nBase = blockIdx.x * 128 + wn * 64;

  const h16* a0p = Wgt + (size_t)oBase * K;
  const h16* a1p = Wgt + (size_t)(oBase + 16) * K;

  v8f acc[2][4] = {};
  for (int k0 = 0; k0 < K; k0 += 32) {
    __builtin_prefetch(a0p + k0 + 64, 0, 0);   // global_prefetch_b8
    v16h a0 = load_A_rowmajor(a0p + k0, K);
    v16h a1 = load_A_rowmajor(a1p + k0, K);
#pragma unroll
    for (int s = 0; s < 4; ++s) {
      v16h bf = load_B_rows(Xb + (size_t)(nBase + s * 16) * K + k0, K);
      acc[0][s] = wmma_f16(a0, bf, acc[0][s]);
      acc[1][s] = wmma_f16(a1, bf, acc[1][s]);
    }
  }

  int mOff = (lane >> 4) << 3;
#pragma unroll
  for (int r = 0; r < 2; ++r)
#pragma unroll
    for (int s = 0; s < 4; ++s) {
      int n = nBase + s * 16 + (lane & 15);
#pragma unroll
      for (int g = 0; g < 8; ++g) {
        int m = oBase + r * 16 + g + mOff;
        size_t idx = (size_t)b * O * N + (size_t)m * N + n;
        float v = acc[r][s][g];
        if (resid) v += resid[idx];
        if (outF)  outF[idx] = v;
        if (outH)  outH[idx] = (h16)v;
      }
    }
}

// ---------------------------------------------------------------------------
// Depthwise 3x3 SAME conv, f16 in/out, f32 weights [C][9]
// ---------------------------------------------------------------------------
__global__ void df_dwconv3x3(const h16* __restrict__ xin,
                             const float* __restrict__ wdw,
                             h16* __restrict__ out, int Cc, int total) {
  int idx = blockIdx.x * 256 + threadIdx.x;
  if (idx >= total) return;
  int wpos = idx & (WW - 1);
  int t = idx >> 7;
  int hpos = t & (HH - 1);
  t >>= 7;
  int c = t % Cc;
  int bc = t;
  const h16* xc = xin + (size_t)bc * NN;
  const float* wk = wdw + c * 9;
  float acc = 0.f;
#pragma unroll
  for (int dy = -1; dy <= 1; ++dy) {
    int yy = hpos + dy;
    if (yy < 0 || yy >= HH) continue;
#pragma unroll
    for (int dx = -1; dx <= 1; ++dx) {
      int xx = wpos + dx;
      if (xx < 0 || xx >= WW) continue;
      acc += wk[(dy + 1) * 3 + (dx + 1)] * (float)xc[yy * WW + xx];
    }
  }
  out[(size_t)bc * NN + hpos * WW + wpos] = (h16)acc;
}

// ---------------------------------------------------------------------------
// Per-row L2 normalization over N (in place).
// ---------------------------------------------------------------------------
__global__ void df_l2norm_rows(h16* __restrict__ data, int usedC, int strideC) {
  int row = blockIdx.x;
  int b = row / usedC;
  int c = row - b * usedC;
  h16* p = data + ((size_t)b * strideC + c) * NN;
  float s = 0.f;
  for (int i = threadIdx.x; i < NN; i += 256) {
    float v = (float)p[i];
    s += v * v;
  }
  __shared__ float red[256];
  red[threadIdx.x] = s;
  __syncthreads();
  for (int st = 128; st > 0; st >>= 1) {
    if (threadIdx.x < st) red[threadIdx.x] += red[threadIdx.x + st];
    __syncthreads();
  }
  float inv = rsqrtf(red[0] + 1e-12f);
  for (int i = threadIdx.x; i < NN; i += 256) p[i] = (h16)((float)p[i] * inv);
}

// ---------------------------------------------------------------------------
// Attention scores: scores[bh][i][j] = sum_n q[i,n]*k[j,n]  (48x48, K=16384)
// 8 waves split-K; ds_add_f32 LDS reduction.
// ---------------------------------------------------------------------------
__global__ __launch_bounds__(256) void df_attn_scores(
    const h16* __restrict__ q,   // [B][384][N], L2-normalized rows
    const h16* __restrict__ kv,  // [B][768][N], k = chans 0..383 (normalized)
    float* __restrict__ scores) {
  int bh = blockIdx.x;
  int b = bh >> 3, h = bh & 7;
  const h16* qb = q  + ((size_t)b * CC + h * CHD) * NN;
  const h16* kb = kv + ((size_t)b * 2 * CC + h * CHD) * NN;

  int wave = threadIdx.x >> 5;
  int kBeg = wave * (NN / 8);
  int kEnd = kBeg + (NN / 8);

  v8f acc[3][3] = {};
  for (int k0 = kBeg; k0 < kEnd; k0 += 32) {
    v16h afr[3], bfr[3];
#pragma unroll
    for (int mi = 0; mi < 3; ++mi)
      afr[mi] = load_A_rowmajor(qb + (size_t)mi * 16 * NN + k0, NN);
#pragma unroll
    for (int nj = 0; nj < 3; ++nj)
      bfr[nj] = load_B_rows(kb + (size_t)nj * 16 * NN + k0, NN);
#pragma unroll
    for (int mi = 0; mi < 3; ++mi)
#pragma unroll
      for (int nj = 0; nj < 3; ++nj)
        acc[mi][nj] = wmma_f16(afr[mi], bfr[nj], acc[mi][nj]);
  }

  __shared__ float red[3][3][16][16];
  for (int i = threadIdx.x; i < 2304; i += 256) ((float*)red)[i] = 0.f;
  __syncthreads();
  int lane = threadIdx.x & 31;
  int mOff = (lane >> 4) << 3;
#pragma unroll
  for (int mi = 0; mi < 3; ++mi)
#pragma unroll
    for (int nj = 0; nj < 3; ++nj)
#pragma unroll
      for (int g = 0; g < 8; ++g)
        atomicAdd(&red[mi][nj][g + mOff][lane & 15], acc[mi][nj][g]);
  __syncthreads();

  float* out = scores + (size_t)bh * 48 * 48;
  for (int i = threadIdx.x; i < 2304; i += 256) {
    int rr = i / 48, cc2 = i - rr * 48;
    out[i] = red[rr >> 4][cc2 >> 4][rr & 15][cc2 & 15];
  }
}

// ---------------------------------------------------------------------------
// Softmax over j (48), pre-scale temp[h]; f16 probs padded to K=64.
// ---------------------------------------------------------------------------
__global__ void df_softmax(const float* __restrict__ scores,
                           const float* __restrict__ temp,
                           h16* __restrict__ probs) {
  int r = blockIdx.x * 256 + threadIdx.x;
  if (r >= BB * HEADS * 48) return;
  int bh = r / 48;
  int h = bh & 7;
  float t = temp[h];
  const float* s = scores + (size_t)r * 48;
  float mx = -3.4e38f;
  float e[48];
  for (int j = 0; j < 48; ++j) { float v = s[j] * t; e[j] = v; mx = fmaxf(mx, v); }
  float sum = 0.f;
  for (int j = 0; j < 48; ++j) { e[j] = __expf(e[j] - mx); sum += e[j]; }
  float inv = 1.f / sum;
  h16* p = probs + (size_t)r * 64;
  for (int j = 0; j < 48; ++j) p[j] = (h16)(e[j] * inv);
  for (int j = 48; j < 64; ++j) p[j] = (h16)0.f;
}

// ---------------------------------------------------------------------------
// attn_out[i,n] = sum_j probs[i,j] * v[j,n]; v read from vT [B][N][384]
// (contiguous B fragments). grid: (N/64, 3 m-tiles, B*HEADS), 1 wave.
// ---------------------------------------------------------------------------
__global__ __launch_bounds__(32) void df_attn_apply(
    const h16* __restrict__ probs,  // [B*HEADS][48][64] (cols 48..63 zero)
    const h16* __restrict__ vT,     // [B][N][384]
    h16* __restrict__ ao) {         // [B][384][N]
  int bh = blockIdx.z;
  int b = bh >> 3, h = bh & 7;
  int mt = blockIdx.y;
  int nBase = blockIdx.x * 64;
  int lane = threadIdx.x & 31;

  const h16* A  = probs + (size_t)bh * 48 * 64 + (size_t)mt * 16 * 64;
  const h16* Vb = vT + (size_t)b * NN * CC + h * CHD;   // + n*384 + j

  v8f acc[4] = {};
#pragma unroll
  for (int k0 = 0; k0 < 64; k0 += 32) {
    v16h afr = load_A_rowmajor(A + k0, 64);
#pragma unroll
    for (int s = 0; s < 4; ++s) {
      // B frag: lane col n, contiguous j run (j>=48 reads junk * zero-prob)
      v16h bf = load_B_rows(Vb + (size_t)(nBase + s * 16) * CC + k0, CC);
      acc[s] = wmma_f16(afr, bf, acc[s]);
    }
  }

  h16* O = ao + ((size_t)b * CC + h * CHD + mt * 16) * NN;
  int mOff = (lane >> 4) << 3;
#pragma unroll
  for (int s = 0; s < 4; ++s)
#pragma unroll
    for (int g = 0; g < 8; ++g)
      O[(size_t)(g + mOff) * NN + nBase + s * 16 + (lane & 15)] = (h16)acc[s][g];
}

// ---------------------------------------------------------------------------
// GELU(exact) gate: g[b][j][n] = gelu(hx[b][j][n]) * hx[b][768+j][n]
// ---------------------------------------------------------------------------
__global__ void df_gelu_gate(const h16* __restrict__ hx, h16* __restrict__ g, int total) {
  int i = blockIdx.x * 256 + threadIdx.x;
  if (i >= total) return;
  int b = i / (HID * NN);
  int rem = i - b * (HID * NN);
  const h16* x1 = hx + (size_t)b * FFN2 * NN + rem;
  const h16* x2 = x1 + (size_t)HID * NN;
  float a = (float)*x1;
  float ge = 0.5f * a * (1.f + erff(a * 0.70710678118f));
  g[i] = (h16)(ge * (float)*x2);
}

// ---------------------------------------------------------------------------
// Host-side launch
// ---------------------------------------------------------------------------
extern "C" void kernel_launch(void* const* d_in, const int* in_sizes, int n_in,
                              void* d_out, int out_size, void* d_ws, size_t ws_size,
                              hipStream_t stream) {
  (void)in_sizes; (void)n_in; (void)out_size; (void)ws_size;

  const float* x       = (const float*)d_in[0];
  const float* y       = (const float*)d_in[1];
  const float* n1_w    = (const float*)d_in[2];
  const float* n1_b    = (const float*)d_in[3];
  const float* n1y_w   = (const float*)d_in[4];
  const float* n1y_b   = (const float*)d_in[5];
  const float* q_w     = (const float*)d_in[6];
  const float* q_dw    = (const float*)d_in[7];
  const float* kv_w    = (const float*)d_in[8];
  const float* kv_dw   = (const float*)d_in[9];
  const float* temp    = (const float*)d_in[10];
  const float* proj_w  = (const float*)d_in[11];
  const float* n2_w    = (const float*)d_in[12];
  const float* n2_b    = (const float*)d_in[13];
  const float* ffn_in_w  = (const float*)d_in[14];
  const float* ffn_dw    = (const float*)d_in[15];
  const float* ffn_out_w = (const float*)d_in[16];
  float* out = (float*)d_out;

  // --- workspace carve (bump allocator, 256B aligned) ---
  char* base = (char*)d_ws;
  size_t off = 0;
  auto alloc = [&](size_t bytes) -> void* {
    void* p = base + off;
    off = (off + bytes + 255) & ~(size_t)255;
    return p;
  };
  const size_t ACT = (size_t)BB * CC * NN;
  h16* wq16   = (h16*)alloc((size_t)CC * CC * 2);
  h16* wkv16  = (h16*)alloc((size_t)2 * CC * CC * 2);
  h16* wpr16  = (h16*)alloc((size_t)CC * CC * 2);
  h16* wfi16  = (h16*)alloc((size_t)FFN2 * CC * 2);
  h16* wfo16  = (h16*)alloc((size_t)CC * HID * 2);
  h16* xn16   = (h16*)alloc(ACT * 2);
  h16* yn16   = (h16*)alloc(ACT * 2);
  h16* xnT    = (h16*)alloc(ACT * 2);          // [B][N][384]
  h16* ynT    = (h16*)alloc(ACT * 2);
  h16* qb16   = (h16*)alloc(ACT * 2);
  h16* kvb16  = (h16*)alloc(ACT * 4);
  h16* q16    = (h16*)alloc(ACT * 2);
  h16* kv16   = (h16*)alloc(ACT * 4);
  h16* vT     = (h16*)alloc(ACT * 2 + 1024);   // [B][N][384] (+pad for j>=48 overread)
  float* scor = (float*)alloc((size_t)BB * HEADS * 48 * 48 * 4);
  h16* prob16 = (h16*)alloc((size_t)BB * HEADS * 48 * 64 * 2);
  h16* ao16   = (h16*)alloc(ACT * 2);
  h16* aoT    = (h16*)alloc(ACT * 2);
  float* x1   = (float*)alloc(ACT * 4);
  h16* x1n16  = (h16*)alloc(ACT * 2);
  h16* x1nT   = (h16*)alloc(ACT * 2);
  h16* hx16   = (h16*)alloc((size_t)BB * FFN2 * NN * 2);
  h16* hdw16  = (h16*)alloc((size_t)BB * FFN2 * NN * 2);
  h16* g16    = (h16*)alloc((size_t)BB * HID * NN * 2);
  h16* gT     = (h16*)alloc((size_t)BB * HID * NN * 2);  // [B][N][768]

  auto cvt = [&](const float* s, h16* d, int n) {
    df_f32_to_f16<<<(n + 255) / 256, 256, 0, stream>>>(s, d, n);
  };
  auto transp = [&](const h16* in, h16* o, int R, int Cl,
                    size_t inBS, size_t outBS) {
    df_transpose16<<<dim3(Cl / 32, R / 32, BB), 256, 0, stream>>>(in, o, R, Cl, inBS, outBS);
  };

  // 1. weights -> f16
  cvt(q_w,       wq16,  CC * CC);
  cvt(kv_w,      wkv16, 2 * CC * CC);
  cvt(proj_w,    wpr16, CC * CC);
  cvt(ffn_in_w,  wfi16, FFN2 * CC);
  cvt(ffn_out_w, wfo16, CC * HID);

  // 2. LayerNorm x, y -> f16, then transpose to N-major
  df_layernorm_f16<<<(BB * NN) / 256, 256, 0, stream>>>(x, n1_w,  n1_b,  xn16, CC, NN);
  df_layernorm_f16<<<(BB * NN) / 256, 256, 0, stream>>>(y, n1y_w, n1y_b, yn16, CC, NN);
  transp(xn16, xnT, CC, NN, (size_t)CC * NN, (size_t)CC * NN);
  transp(yn16, ynT, CC, NN, (size_t)CC * NN, (size_t)CC * NN);

  // 3. q / kv channel GEMMs (WMMA)
  df_gemm_wmma_t<<<dim3(NN / 128, CC / 128, BB), 256, 0, stream>>>(
      wq16, xnT, nullptr, nullptr, qb16, CC, CC, NN);
  df_gemm_wmma_t<<<dim3(NN / 128, (2 * CC) / 128, BB), 256, 0, stream>>>(
      wkv16, ynT, nullptr, nullptr, kvb16, 2 * CC, CC, NN);

  // 4. depthwise 3x3
  df_dwconv3x3<<<(BB * CC * NN) / 256, 256, 0, stream>>>(qb16, q_dw, q16, CC, BB * CC * NN);
  df_dwconv3x3<<<(BB * 2 * CC * NN) / 256, 256, 0, stream>>>(kvb16, kv_dw, kv16, 2 * CC,
                                                             BB * 2 * CC * NN);

  // 5. L2 normalize q rows and k rows
  df_l2norm_rows<<<BB * CC, 256, 0, stream>>>(q16, CC, CC);
  df_l2norm_rows<<<BB * CC, 256, 0, stream>>>(kv16, CC, 2 * CC);

  // 6. transpose v half for attn_apply
  transp(kv16 + (size_t)CC * NN, vT, CC, NN, (size_t)2 * CC * NN, (size_t)CC * NN);

  // 7. attention
  df_attn_scores<<<BB * HEADS, 256, 0, stream>>>(q16, kv16, scor);
  df_softmax<<<(BB * HEADS * 48 + 255) / 256, 256, 0, stream>>>(scor, temp, prob16);
  df_attn_apply<<<dim3(NN / 64, 3, BB * HEADS), 32, 0, stream>>>(prob16, vT, ao16);
  transp(ao16, aoT, CC, NN, (size_t)CC * NN, (size_t)CC * NN);

  // 8. proj GEMM + residual(x) -> x1 f32
  df_gemm_wmma_t<<<dim3(NN / 128, CC / 128, BB), 256, 0, stream>>>(
      wpr16, aoT, x, x1, nullptr, CC, CC, NN);

  // 9. LayerNorm2 + transpose
  df_layernorm_f16<<<(BB * NN) / 256, 256, 0, stream>>>(x1, n2_w, n2_b, x1n16, CC, NN);
  transp(x1n16, x1nT, CC, NN, (size_t)CC * NN, (size_t)CC * NN);

  // 10. FFN in GEMM (1536 x 384)
  df_gemm_wmma_t<<<dim3(NN / 128, FFN2 / 128, BB), 256, 0, stream>>>(
      wfi16, x1nT, nullptr, nullptr, hx16, FFN2, CC, NN);

  // 11. FFN depthwise + GELU gate
  df_dwconv3x3<<<(BB * FFN2 * NN) / 256, 256, 0, stream>>>(hx16, ffn_dw, hdw16, FFN2,
                                                           BB * FFN2 * NN);
  df_gelu_gate<<<(BB * HID * NN) / 256, 256, 0, stream>>>(hdw16, g16, BB * HID * NN);
  transp(g16, gT, HID, NN, (size_t)HID * NN, (size_t)HID * NN);

  // 12. FFN out GEMM (384 x 768) + residual(x1) -> d_out
  df_gemm_wmma_t<<<dim3(NN / 128, CC / 128, BB), 256, 0, stream>>>(
      wfo16, gT, x1, out, nullptr, CC, HID, NN);
}
